// SearchTransfer_3444563772133
// MI455X (gfx1250) — matrix operand order repaired and checked by reading, not compiled, not run on hardware
//
#include <hip/hip_runtime.h>
#include <hip/hip_bf16.h>

typedef __attribute__((ext_vector_type(16))) _Float16 v16h;
typedef __attribute__((ext_vector_type(8)))  _Float16 v8h;
typedef __attribute__((ext_vector_type(8)))  float    v8f;

#define HW3   48
#define NPOS  2304      // 48*48
#define C3    256
#define FDIM  2304      // 256*9
#define BATCH 4

// LDS Q-tile: 32 m-rows, padded row stride (2312 halves -> 4624B, stride%256B!=0
// so 16 lanes x ds_load_b128 land on distinct bank groups)
#define MROWS   32
#define LSTRIDE 2312

// Flip to 0 if the assembler rejects the async mnemonic.
#define ASYNC_FILL 1

// ---------------------------------------------------------------------------
// 1) per-position sum over channels of x^2  (q: z=0, k: z=1)
// ---------------------------------------------------------------------------
__global__ void sqsum_kernel(const float* __restrict__ q, const float* __restrict__ k,
                             float* __restrict__ sq) {
  const int p = blockIdx.x * 128 + threadIdx.x;        // 18*128 = 2304
  const int b = blockIdx.y;
  const float* src = (blockIdx.z ? k : q) + (size_t)b * C3 * NPOS + p;
  float s = 0.f;
  for (int c = 0; c < C3; ++c) {
    float v = src[(size_t)c * NPOS];
    s += v * v;
  }
  sq[blockIdx.z * (BATCH * NPOS) + b * NPOS + p] = s;
}

// ---------------------------------------------------------------------------
// 2) 3x3 zero-padded box sum of sq -> 1/max(sqrt(.),EPS)
// ---------------------------------------------------------------------------
__global__ void invnorm_kernel(const float* __restrict__ sq, float* __restrict__ rn) {
  const int p = blockIdx.x * 128 + threadIdx.x;
  const int b = blockIdx.y;
  const int t = blockIdx.z;
  const int y = p / HW3, x = p % HW3;
  const float* s = sq + t * (BATCH * NPOS) + b * NPOS;
  float acc = 0.f;
  for (int dy = -1; dy <= 1; ++dy)
    for (int dx = -1; dx <= 1; ++dx) {
      int yy = y + dy, xx = x + dx;
      if (yy >= 0 && yy < HW3 && xx >= 0 && xx < HW3) acc += s[yy * HW3 + xx];
    }
  rn[t * (BATCH * NPOS) + b * NPOS + p] = 1.0f / fmaxf(sqrtf(acc), 1e-12f);
}

// ---------------------------------------------------------------------------
// 3) pack normalized unfold rows as f16, row-major [pos][feature]
// ---------------------------------------------------------------------------
__global__ void pack_kernel(const float* __restrict__ q, const float* __restrict__ k,
                            const float* __restrict__ rn,
                            _Float16* __restrict__ Qn, _Float16* __restrict__ Kn) {
  const int pos = blockIdx.x;
  const int b   = blockIdx.y;
  const int t   = blockIdx.z;
  const float* src = t ? k : q;
  _Float16* dst    = t ? Kn : Qn;
  const float r = rn[t * (BATCH * NPOS) + b * NPOS + pos];
  const int y = pos / HW3, x = pos % HW3;
  _Float16* row = dst + ((size_t)b * NPOS + pos) * FDIM;
  for (int f = threadIdx.x; f < FDIM; f += 256) {
    int c = f / 9, ij = f % 9, i = ij / 3, j = ij % 3;
    int yy = y + i - 1, xx = x + j - 1;
    float v = (yy >= 0 && yy < HW3 && xx >= 0 && xx < HW3)
                  ? src[((size_t)(b * C3 + c)) * NPOS + yy * HW3 + xx]
                  : 0.f;
    row[f] = (_Float16)(v * r);
  }
}

// ---------------------------------------------------------------------------
// 4) relevance GEMM on WMMA + streaming max/argmax over L
//    block = (b, m-tile of 32): Q tile staged in LDS (async) and shared by all
//    8 waves; each wave streams its share of the 144 l-tiles from global and
//    computes a 16l x 32m register tile (2 accumulators per A operand).
// ---------------------------------------------------------------------------
__global__ __launch_bounds__(256)
void relevance_wmma_kernel(const _Float16* __restrict__ Kn, const _Float16* __restrict__ Qn,
                           float* __restrict__ S_out, int* __restrict__ idx_out) {
  __shared__ __align__(16) _Float16 qtile[MROWS * LSTRIDE];   // ~145 KB
  __shared__ float svals[8][32];
  __shared__ int   slidx[8][32];

  const int b    = blockIdx.y;
  const int m0   = blockIdx.x << 5;                 // 32 m per block
  const int wave = threadIdx.x >> 5;
  const int lane = threadIdx.x & 31;
  const int g    = lane >> 4;                       // half-wave group
  const int r    = lane & 15;                       // row/col within tile

  // ---- stage Q tile (32 rows x 2304 halves) into LDS --------------------
  // 16B chunks: 32 rows x 288 chunks = 9216, 36 per thread
  for (int t = threadIdx.x; t < MROWS * (FDIM / 8); t += 256) {
    const int row = t / (FDIM / 8);
    const int ch  = t % (FDIM / 8);
    const _Float16* gsrc = Qn + ((size_t)b * NPOS + m0 + row) * FDIM + ch * 8;
    _Float16* ldst = qtile + row * LSTRIDE + ch * 8;
#if ASYNC_FILL
    // VDST VGPR = LDS byte offset (low 32 bits of generic LDS address),
    // VADDR = 64-bit global address.  Tracked on ASYNCcnt.
    unsigned int      loff = (unsigned int)(unsigned long long)(uintptr_t)ldst;
    unsigned long long ga  = (unsigned long long)(uintptr_t)gsrc;
    asm volatile("global_load_async_to_lds_b128 %0, %1, off"
                 :: "v"(loff), "v"(ga) : "memory");
#else
    *(v8h*)ldst = *(const v8h*)gsrc;
#endif
  }
#if ASYNC_FILL
  asm volatile("s_wait_asynccnt 0" ::: "memory");
#endif
  __syncthreads();

  // B operands out of LDS: lane holds column m0+sub*16+r, K range g*16..+15
  const _Float16* Brow0 = qtile + (size_t)r * LSTRIDE + g * 16;
  const _Float16* Brow1 = qtile + (size_t)(16 + r) * LSTRIDE + g * 16;

  float best0 = -3.0e38f, best1 = -3.0e38f;
  int bestl0 = 0, bestl1 = 0;

  for (int lt = wave; lt < 144; lt += 8) {
    const int l0 = lt << 4;
    // A operand (16x32, M x K): lane holds row l0+r; halves f0+g*16, f0+g*16+8
    const _Float16* Krow = Kn + ((size_t)b * NPOS + l0 + r) * FDIM + g * 16;
    if (lt + 8 < 144)
      __builtin_prefetch(Krow + (size_t)128 * FDIM, 0, 3);   // next l-tile row

    v8f acc0 = {}, acc1 = {};
    for (int f0 = 0; f0 < FDIM; f0 += 32) {
      v8h a_lo = *(const v8h*)(Krow + f0);
      v8h a_hi = *(const v8h*)(Krow + f0 + 8);
      v16h A = __builtin_shufflevector(a_lo, a_hi,
                                       0,1,2,3,4,5,6,7,8,9,10,11,12,13,14,15);
      v8h b0l = *(const v8h*)(Brow0 + f0);
      v8h b0h = *(const v8h*)(Brow0 + f0 + 8);
      v8h b1l = *(const v8h*)(Brow1 + f0);
      v8h b1h = *(const v8h*)(Brow1 + f0 + 8);
      v16h B0 = __builtin_shufflevector(b0l, b0h,
                                        0,1,2,3,4,5,6,7,8,9,10,11,12,13,14,15);
      v16h B1 = __builtin_shufflevector(b1l, b1h,
                                        0,1,2,3,4,5,6,7,8,9,10,11,12,13,14,15);
      acc0 = __builtin_amdgcn_wmma_f32_16x16x32_f16(false, A, false, B0,
                                                    (short)0, acc0, false, false);
      acc1 = __builtin_amdgcn_wmma_f32_16x16x32_f16(false, A, false, B1,
                                                    (short)0, acc1, false, false);
    }
    // C layout: VGPR t -> row l0 + g*8 + t, column = lane&15 (per sub-tile)
    {
      float v = acc0[0]; int li = l0 + g * 8;
#pragma unroll
      for (int t2 = 1; t2 < 8; ++t2) {
        float w = acc0[t2]; int lw = l0 + g * 8 + t2;
        if (w > v) { v = w; li = lw; }
      }
      float vo = __shfl_xor(v, 16, 32);
      int   lo = __shfl_xor(li, 16, 32);
      if (vo > v || (vo == v && lo < li)) { v = vo; li = lo; }
      if (v > best0 || (v == best0 && li < bestl0)) { best0 = v; bestl0 = li; }
    }
    {
      float v = acc1[0]; int li = l0 + g * 8;
#pragma unroll
      for (int t2 = 1; t2 < 8; ++t2) {
        float w = acc1[t2]; int lw = l0 + g * 8 + t2;
        if (w > v) { v = w; li = lw; }
      }
      float vo = __shfl_xor(v, 16, 32);
      int   lo = __shfl_xor(li, 16, 32);
      if (vo > v || (vo == v && lo < li)) { v = vo; li = lo; }
      if (v > best1 || (v == best1 && li < bestl1)) { best1 = v; bestl1 = li; }
    }
  }

  if (lane < 16) {
    svals[wave][lane]      = best0;  slidx[wave][lane]      = bestl0;
    svals[wave][lane + 16] = best1;  slidx[wave][lane + 16] = bestl1;
  }
  __syncthreads();

  if (threadIdx.x < 32) {
    float v = svals[0][threadIdx.x];
    int  li = slidx[0][threadIdx.x];
    for (int w2 = 1; w2 < 8; ++w2) {
      float vv = svals[w2][threadIdx.x];
      int   ll = slidx[w2][threadIdx.x];
      if (vv > v || (vv == v && ll < li)) { v = vv; li = ll; }
    }
    S_out[(size_t)b * NPOS + m0 + threadIdx.x] = v;
    idx_out[b * NPOS + m0 + threadIdx.x] = li;
  }
}

// ---------------------------------------------------------------------------
// 5) fused take_along_axis + fold + cnt normalization for value pyramid
//    S = stride/scale (1,2,4); kernel size = 3S, pad = S, out = 48S x 48S
// ---------------------------------------------------------------------------
template <int S>
__global__ void fold_gather_kernel(const float* __restrict__ val, const int* __restrict__ idx,
                                   float* __restrict__ out, int C) {
  const int HW = 48 * S;
  const long long total = (long long)BATCH * C * HW * HW;
  long long n = (long long)blockIdx.x * blockDim.x + threadIdx.x;
  if (n >= total) return;
  int x = (int)(n % HW);
  int y = (int)((n / HW) % HW);
  int c = (int)((n / ((long long)HW * HW)) % C);
  int b = (int)(n / ((long long)HW * HW * C));

  const int yq = y / S, yr = y % S;
  const int xq = x / S, xr = x % S;

  float sum = 0.f;
  int cnt = 0;
  for (int ti = 0; ti < 3; ++ti) {
    const int i  = yr + S * ti;
    const int ho = yq + 1 - ti;
    if (ho < 0 || ho >= 48) continue;
    for (int tj = 0; tj < 3; ++tj) {
      const int j  = xr + S * tj;
      const int wo = xq + 1 - tj;
      if (wo < 0 || wo >= 48) continue;
      ++cnt;
      const int l  = idx[b * NPOS + ho * 48 + wo];
      const int vy = S * (l / 48) + i - S;
      const int vx = S * (l % 48) + j - S;
      if (vy >= 0 && vy < HW && vx >= 0 && vx < HW)
        sum += val[(((size_t)b * C + c) * HW + vy) * HW + vx];
    }
  }
  out[n] = sum / (float)cnt;
}

// ---------------------------------------------------------------------------
extern "C" void kernel_launch(void* const* d_in, const int* in_sizes, int n_in,
                              void* d_out, int out_size, void* d_ws, size_t ws_size,
                              hipStream_t stream) {
  (void)in_sizes; (void)n_in; (void)out_size; (void)ws_size;

  const float* query = (const float*)d_in[0];   // (4,256,48,48)
  const float* key   = (const float*)d_in[1];   // (4,256,48,48)
  const float* v1    = (const float*)d_in[2];   // (4, 64,192,192)
  const float* v2    = (const float*)d_in[3];   // (4,128, 96, 96)
  const float* v3    = (const float*)d_in[4];   // (4,256, 48, 48)
  // d_in[5] (cl_ref), d_in[6] (dr_img) are unused by the reference.

  // workspace carve
  const size_t MATB = (size_t)BATCH * NPOS * FDIM * sizeof(_Float16); // 42,467,328 B
  char* w = (char*)d_ws;
  _Float16* Qn = (_Float16*)w;            w += MATB;
  _Float16* Kn = (_Float16*)w;            w += MATB;
  float* sq    = (float*)w;               w += 2 * BATCH * NPOS * sizeof(float);
  float* rn    = (float*)w;               w += 2 * BATCH * NPOS * sizeof(float);
  int*   idx   = (int*)w;                 w += BATCH * NPOS * sizeof(int);

  float* out = (float*)d_out;
  float* S_out  = out;                       //     9,216
  float* T3_out = out + 9216;                // 2,359,296
  float* T2_out = out + 2368512;             // 4,718,592
  float* T1_out = out + 7087104;             // 9,437,184

  dim3 gp(18, BATCH, 2);
  sqsum_kernel<<<gp, 128, 0, stream>>>(query, key, sq);
  invnorm_kernel<<<gp, 128, 0, stream>>>(sq, rn);

  dim3 gpk(NPOS, BATCH, 2);
  pack_kernel<<<gpk, 256, 0, stream>>>(query, key, rn, Qn, Kn);

  dim3 gr(NPOS / 32, BATCH);                 // 72 x 4 blocks
  relevance_wmma_kernel<<<gr, 256, 0, stream>>>(Kn, Qn, S_out, idx);

  fold_gather_kernel<1><<<(2359296 + 255) / 256, 256, 0, stream>>>(v3, idx, T3_out, 256);
  fold_gather_kernel<2><<<(4718592 + 255) / 256, 256, 0, stream>>>(v2, idx, T2_out, 128);
  fold_gather_kernel<4><<<(9437184 + 255) / 256, 256, 0, stream>>>(v1, idx, T1_out, 64);
}
